// NMTDecoder_75247827026078
// MI455X (gfx1250) — matrix operand order repaired and checked by reading, not compiled, not run on hardware
//
#include <hip/hip_runtime.h>
#include <stdint.h>

#define BB   32      // batch
#define TT   32      // time steps
#define HIDN 256
#define EMBN 256
#define LENC 512     // encoder length
#define KIN  512     // EMB + HID
#define G3   768     // 3 * HID
#define VOC  32000
#define NBLK 128     // vocab rows per classifier workgroup

typedef __attribute__((ext_vector_type(16))) __bf16       v16bf;
typedef __attribute__((ext_vector_type(8)))  float        v8f;
typedef __attribute__((ext_vector_type(4)))  unsigned int u32x4;
typedef __attribute__((ext_vector_type(8)))  int          i32x8;
typedef __attribute__((ext_vector_type(4)))  int          i32x4;

union Frag16 { v16bf v; u32x4 q[2]; };

// A-matrix 16x32 bf16 fragment (ISA 7.12.2): lane<16 holds row M=laneN,
// K = kbase+0..7 and kbase+16..23; lane>=16 holds K = kbase+8..15 and +24..31.
__device__ __forceinline__ v16bf load_a_frag(const __bf16* base, int row, int rowStride,
                                             int kbase, int hiHalf) {
  Frag16 f;
  const __bf16* p = base + (long)row * rowStride + kbase + (hiHalf ? 8 : 0);
  f.q[0] = *(const u32x4*)(p);
  f.q[1] = *(const u32x4*)(p + 16);
  return f.v;
}

// B-matrix 32x16 bf16 fragment: lanes 0-15 hold K = kbase+0..7 and kbase+16..23
// for column N=laneN; lanes 16-31 hold K = kbase+8..15 and kbase+24..31.
__device__ __forceinline__ v16bf load_b_frag(const __bf16* base, int row, int rowStride,
                                             int kbase, int hiHalf) {
  Frag16 f;
  const __bf16* p = base + (long)row * rowStride + kbase + (hiHalf ? 8 : 0);
  f.q[0] = *(const u32x4*)(p);
  f.q[1] = *(const u32x4*)(p + 16);
  return f.v;
}

// ---------------------------------------------------------------------------
// fp32 -> bf16 weight conversion (grid-stride)
// ---------------------------------------------------------------------------
__global__ void cvt_bf16_kernel(const float* __restrict__ src,
                                __bf16* __restrict__ dst, int n) {
  int i = blockIdx.x * blockDim.x + threadIdx.x;
  int stride = gridDim.x * blockDim.x;
  for (; i < n; i += stride) dst[i] = (__bf16)src[i];
}

// ---------------------------------------------------------------------------
// h0 = initial_hidden @ W_hm^T + b_hm   ([32,256] x [256,256]) — tiny, VALU
// ---------------------------------------------------------------------------
__global__ void h0_kernel(const float* __restrict__ hinit,
                          const float* __restrict__ Whm,
                          const float* __restrict__ bhm,
                          float* __restrict__ h0) {
  int b = blockIdx.x;       // 32 blocks
  int j = threadIdx.x;      // 256 threads
  float s = bhm[j];
  const float* hv = hinit + b * HIDN;
  const float* wr = Whm + j * HIDN;
  for (int d = 0; d < HIDN; ++d) s += hv[d] * wr[d];
  h0[b * HIDN + j] = s;
}

// ---------------------------------------------------------------------------
// Sequential GRU + attention recurrence. One workgroup, 1024 threads = 32
// wave32 waves. Batch (B=32) is the WMMA M dimension. Emits pred[t*32+b][0:512]
// = concat(ctx, h_new) in bf16 for the classifier GEMM.
// LDS: 96KB gate staging (aliased by softmax scores [0,64K) + f32 h [64K,96K))
//      + 32KB rnn_in + 16KB h(bf16) = 144KB.
// ---------------------------------------------------------------------------
__global__ __launch_bounds__(1024) void gru_attn_kernel(
    const float*  __restrict__ enc,    // [32][512][256] f32
    const int*    __restrict__ tgt,    // [32][32]
    const float*  __restrict__ Wemb,   // [32000][256] f32
    const __bf16* __restrict__ Wih,    // [768][512] bf16
    const __bf16* __restrict__ Whh,    // [768][256] bf16
    const float*  __restrict__ bih,    // [768]
    const float*  __restrict__ bhh,    // [768]
    const float*  __restrict__ h0,     // [32][256] f32 (ws)
    __bf16*       __restrict__ pred)   // [1024][512] bf16 (ws)
{
  __shared__ __attribute__((aligned(16))) char   s_raw[2 * BB * G3 * 2];
  __shared__ __attribute__((aligned(16))) __bf16 s_in[BB * KIN];   // 32KB rnn_in
  __shared__ __attribute__((aligned(16))) __bf16 s_hb[BB * HIDN];  // 16KB hidden bf16

  __bf16* s_gi = (__bf16*)s_raw;
  __bf16* s_gh = s_gi + BB * G3;
  float*  s_sc = (float*)s_raw;                  // scores: bytes [0, 64K)
  float*  s_hf = (float*)(s_raw + 64 * 1024);    // f32 hidden: bytes [64K, 96K)

  const int tid    = threadIdx.x;
  const int wave   = tid >> 5;
  const int lane   = tid & 31;
  const int laneN  = lane & 15;
  const int hiHalf = lane >> 4;

  for (int i = tid; i < BB * HIDN; i += 1024) {
    s_hb[i] = (__bf16)h0[i];
    int b = i >> 8, d = i & 255;
    s_in[b * KIN + EMBN + d] = (__bf16)0.0f;
  }
  __syncthreads();

  for (int t = 0; t < TT; ++t) {
    // ---- Phase A: embedding gather (padding_idx=0 -> zero row) ----
    for (int i = tid; i < BB * EMBN; i += 1024) {
      int b = i >> 8, e = i & 255;
      int y = tgt[b * TT + t];
      float xv = (y == 0) ? 0.0f : Wemb[(long)y * EMBN + e];
      s_in[b * KIN + e] = (__bf16)xv;
    }
    __syncthreads();

    // ---- Phase B: gi = rnn_in @ Wih^T, gh = h @ Whh^T via WMMA bf16 ----
    for (int q = 0; q < 3; ++q) {
      int tileIdx = wave * 3 + q;    // 2 m-tiles x 48 n-tiles = 96 tiles
      int mt = tileIdx & 1;
      int nt = tileIdx >> 1;

      v8f acc = {};
      for (int ks = 0; ks < KIN / 32; ++ks) {
        v16bf a  = load_a_frag(s_in, mt * 16 + laneN, KIN, ks * 32, hiHalf);
        v16bf bm = load_b_frag(Wih,  nt * 16 + laneN, KIN, ks * 32, hiHalf);
        acc = __builtin_amdgcn_wmma_f32_16x16x32_bf16(false, a, false, bm,
                                                      (short)0, acc, false, false);
      }
      for (int r = 0; r < 8; ++r)
        s_gi[(mt * 16 + r + hiHalf * 8) * G3 + nt * 16 + laneN] = (__bf16)acc[r];

      v8f acch = {};
      for (int ks = 0; ks < HIDN / 32; ++ks) {
        v16bf a  = load_a_frag(s_hb, mt * 16 + laneN, HIDN, ks * 32, hiHalf);
        v16bf bm = load_b_frag(Whh,  nt * 16 + laneN, HIDN, ks * 32, hiHalf);
        acch = __builtin_amdgcn_wmma_f32_16x16x32_bf16(false, a, false, bm,
                                                       (short)0, acch, false, false);
      }
      for (int r = 0; r < 8; ++r)
        s_gh[(mt * 16 + r + hiHalf * 8) * G3 + nt * 16 + laneN] = (__bf16)acch[r];
    }
    __syncthreads();

    // ---- Phase C: GRU gate fusion -> h_new ----
    for (int i = tid; i < BB * HIDN; i += 1024) {
      int b = i >> 8, j = i & 255;
      float ir = (float)s_gi[b * G3 + j]            + bih[j];
      float iz = (float)s_gi[b * G3 + HIDN + j]     + bih[HIDN + j];
      float in = (float)s_gi[b * G3 + 2 * HIDN + j] + bih[2 * HIDN + j];
      float hr = (float)s_gh[b * G3 + j]            + bhh[j];
      float hz = (float)s_gh[b * G3 + HIDN + j]     + bhh[HIDN + j];
      float hn = (float)s_gh[b * G3 + 2 * HIDN + j] + bhh[2 * HIDN + j];
      float r = 1.0f / (1.0f + __expf(-(ir + hr)));
      float z = 1.0f / (1.0f + __expf(-(iz + hz)));
      float n = tanhf(in + r * hn);
      float hnew = (1.0f - z) * n + z * (float)s_hb[i];
      s_hb[i] = (__bf16)hnew;
      pred[((long)(t * BB + b)) * 512 + HIDN + j] = (__bf16)hnew;
    }
    __syncthreads();

    // ---- Phase C2: stage h_new as f32 in the spare gate-staging LDS ----
    for (int i = tid; i < BB * HIDN; i += 1024)
      s_hf[i] = (float)s_hb[i];
    __syncthreads();

    // ---- Phase D: dot-product attention + softmax; one wave per batch ----
    {
      int b = wave;
      float* sc = s_sc + b * LENC;
      const float* hb = s_hf + b * HIDN;
      const float* Eb = enc + (long)b * LENC * HIDN;

      // scores: each lane owns rows l = lane + 32*i; h chunked into registers
      float acc[16];
#pragma unroll
      for (int i = 0; i < 16; ++i) acc[i] = 0.0f;
      for (int d0 = 0; d0 < HIDN; d0 += 8) {
        float hr[8];
#pragma unroll
        for (int j = 0; j < 8; ++j) hr[j] = hb[d0 + j];
#pragma unroll
        for (int i = 0; i < 16; ++i) {
          const float4* rp = (const float4*)(Eb + (long)(lane + i * 32) * HIDN + d0);
          float4 e0 = rp[0], e1 = rp[1];
          acc[i] += e0.x * hr[0] + e0.y * hr[1] + e0.z * hr[2] + e0.w * hr[3]
                  + e1.x * hr[4] + e1.y * hr[5] + e1.z * hr[6] + e1.w * hr[7];
        }
      }
      float lmax = -3.0e38f;
#pragma unroll
      for (int i = 0; i < 16; ++i) lmax = fmaxf(lmax, acc[i]);
      for (int off = 16; off > 0; off >>= 1)
        lmax = fmaxf(lmax, __shfl_xor(lmax, off, 32));

      float lsum = 0.0f;
#pragma unroll
      for (int i = 0; i < 16; ++i) {
        float e = __expf(acc[i] - lmax);
        sc[lane + i * 32] = e;
        lsum += e;
      }
      for (int off = 16; off > 0; off >>= 1)
        lsum += __shfl_xor(lsum, off, 32);
      float inv = 1.0f / lsum;
      __syncwarp();   // sc[] visibility within the wave before ctx pass

      // ctx: lane owns d = lane + 32*i; softmax weights chunked from LDS
      float cacc[8];
#pragma unroll
      for (int i = 0; i < 8; ++i) cacc[i] = 0.0f;
      for (int l0 = 0; l0 < LENC; l0 += 8) {
        float scr[8];
#pragma unroll
        for (int j = 0; j < 8; ++j) scr[j] = sc[l0 + j];
#pragma unroll
        for (int j = 0; j < 8; ++j) {
          const float* ep = Eb + (long)(l0 + j) * HIDN + lane;
#pragma unroll
          for (int i = 0; i < 8; ++i)
            cacc[i] += ep[i * 32] * scr[j];
        }
      }
#pragma unroll
      for (int i = 0; i < 8; ++i) {
        int d = lane + i * 32;
        __bf16 cb = (__bf16)(cacc[i] * inv);
        s_in[b * KIN + EMBN + d] = cb;                     // ctx for next step
        pred[((long)(t * BB + b)) * 512 + d] = cb;         // pred = [ctx, h]
      }
    }
    __syncthreads();
  }
}

// ---------------------------------------------------------------------------
// Classifier GEMM: logits[1024][32000] = pred[1024][512] @ Wcls^T + b_cls.
// One WG per 128-row W_cls block; block is DMA'd into LDS by the Tensor Data
// Mover (tensor_load_to_lds + s_wait_tensorcnt). Each wave holds 8 n-tile
// accumulators so every pred A-fragment feeds 8 WMMAs (8x less L2 traffic).
// ---------------------------------------------------------------------------
__global__ __launch_bounds__(256) void cls_gemm_kernel(
    const __bf16* __restrict__ pred,   // [1024][512] bf16
    const __bf16* __restrict__ Wc,     // [32000][512] bf16
    const float*  __restrict__ bcls,   // [32000]
    float*        __restrict__ out)    // [32][32][32000] f32
{
  __shared__ __attribute__((aligned(16))) __bf16 s_w[NBLK * 512];  // 128KB
  const int nb  = blockIdx.x;          // 0..249
  const int tid = threadIdx.x;

  // ---- TDM: DMA the 128x512 bf16 tile (128KB) from global into LDS ----
  if ((tid >> 5) == 0) {               // wave 0 issues the descriptor
    unsigned lds_base = (unsigned)(uintptr_t)(&s_w[0]);
    unsigned long long ga =
        (unsigned long long)(uintptr_t)(Wc + (size_t)nb * NBLK * 512);

    u32x4 g0;
    g0[0] = 1u;                                   // count=1 (valid user D#)
    g0[1] = lds_base;                             // lds_addr
    g0[2] = (unsigned)(ga & 0xffffffffu);         // global_addr[31:0]
    g0[3] = (unsigned)((ga >> 32) & 0x1ffffffu)   // global_addr[56:32]
          | 0x80000000u;                          // type=2 ("image")

    i32x8 g1;
    g1[0] = (int)(1u << 16);                      // data_size=1 (2 bytes)
    g1[1] = (int)(512u << 16);                    // tensor_dim0 = 512 (lo16)
    g1[2] = (int)((32000u & 0xffffu) << 16);      // tensor_dim0 hi=0 | tensor_dim1 lo
    g1[3] = (int)((32000u >> 16) | (512u << 16)); // tensor_dim1 hi | tile_dim0=512
    g1[4] = NBLK;                                 // tile_dim1=128, tile_dim2=0
    g1[5] = 512;                                  // tensor_dim0_stride = 512
    g1[6] = 0;
    g1[7] = 0;

    i32x4 z4 = {0, 0, 0, 0};                      // 2D tensor: groups 2/3 unused
#if defined(__clang_major__) && (__clang_major__ >= 23)
    i32x8 z8 = {0, 0, 0, 0, 0, 0, 0, 0};
    __builtin_amdgcn_tensor_load_to_lds(g0, g1, z4, z4, z8, 0);
#else
    __builtin_amdgcn_tensor_load_to_lds(g0, g1, z4, z4, 0);
#endif
    __builtin_amdgcn_s_wait_tensorcnt(0);
  }
  __syncthreads();

  const int wave   = tid >> 5;
  const int lane   = tid & 31;
  const int laneN  = lane & 15;
  const int hiHalf = lane >> 4;

  for (int mq = 0; mq < 8; ++mq) {
    const int mbase = (wave * 8 + mq) * 16;       // 64 m-tiles over 8 waves
    v8f zero = {};
    v8f acc[8];
#pragma unroll
    for (int nt = 0; nt < 8; ++nt) acc[nt] = zero;

    for (int ks = 0; ks < 16; ++ks) {
      v16bf a = load_a_frag(pred, mbase + laneN, 512, ks * 32, hiHalf);
#pragma unroll
      for (int nt = 0; nt < 8; ++nt) {
        v16bf bm = load_b_frag(s_w, nt * 16 + laneN, 512, ks * 32, hiHalf);
        acc[nt] = __builtin_amdgcn_wmma_f32_16x16x32_bf16(false, a, false, bm,
                                                          (short)0, acc[nt],
                                                          false, false);
      }
    }

#pragma unroll
    for (int nt = 0; nt < 8; ++nt) {
      int v = nb * NBLK + nt * 16 + laneN;
      float bias = bcls[v];
#pragma unroll
      for (int r = 0; r < 8; ++r) {
        int m = mbase + r + hiHalf * 8;
        int b = m & 31, tstep = m >> 5;           // pred row = t*32 + b
        out[((long)(b * TT + tstep)) * VOC + v] = acc[nt][r] + bias;
      }
    }
  }
}

// ---------------------------------------------------------------------------
extern "C" void kernel_launch(void* const* d_in, const int* in_sizes, int n_in,
                              void* d_out, int out_size, void* d_ws, size_t ws_size,
                              hipStream_t stream) {
  (void)in_sizes; (void)n_in; (void)out_size; (void)ws_size;

  const float* enc   = (const float*)d_in[0];
  const float* hinit = (const float*)d_in[1];
  const int*   tgt   = (const int*)  d_in[2];
  const float* Wemb  = (const float*)d_in[3];
  const float* Wih   = (const float*)d_in[4];
  const float* Whh   = (const float*)d_in[5];
  const float* bih   = (const float*)d_in[6];
  const float* bhh   = (const float*)d_in[7];
  const float* Whm   = (const float*)d_in[8];
  const float* bhm   = (const float*)d_in[9];
  const float* Wcls  = (const float*)d_in[10];
  const float* bcls  = (const float*)d_in[11];
  float* out = (float*)d_out;

  // workspace layout
  char* ws = (char*)d_ws;
  float*  ws_h    = (float*)(ws);                                   //  32 KB
  __bf16* ws_pred = (__bf16*)(ws + 32768);                          //   1 MB
  __bf16* ws_Wih  = (__bf16*)(ws + 32768 + (size_t)1024*512*2);
  __bf16* ws_Whh  = (__bf16*)((char*)ws_Wih + (size_t)G3*KIN*2);
  __bf16* ws_Wcls = (__bf16*)((char*)ws_Whh + (size_t)G3*HIDN*2);   //  32 MB

  cvt_bf16_kernel<<<512, 256, 0, stream>>>(Wih, ws_Wih, G3 * KIN);
  cvt_bf16_kernel<<<256, 256, 0, stream>>>(Whh, ws_Whh, G3 * HIDN);
  cvt_bf16_kernel<<<4096, 256, 0, stream>>>(Wcls, ws_Wcls, VOC * KIN);
  h0_kernel<<<BB, HIDN, 0, stream>>>(hinit, Whm, bhm, ws_h);

  gru_attn_kernel<<<1, 1024, 0, stream>>>(enc, tgt, Wemb, ws_Wih, ws_Whh,
                                          bih, bhh, ws_h, ws_pred);

  cls_gemm_kernel<<<VOC / NBLK, 256, 0, stream>>>(ws_pred, ws_Wcls, bcls, out);
}
